// LinearRetriever_15281493639180
// MI455X (gfx1250) — matrix operand order repaired and checked by reading, not compiled, not run on hardware
//
#include <hip/hip_runtime.h>

// Problem constants
#define XROWS 16384          // 64 * 256
#define ROWS  20480          // + 64 * 64 y-rows
#define DIN   512
#define EDIM  200
#define EP    224            // projected row stride (bf16 elems), padded for K%32
#define LP    232            // LDS row stride (bf16 elems): bank-conflict padding

#define USE_ASYNC_LDS 1

typedef __attribute__((ext_vector_type(16))) __bf16         v16bf;
typedef __attribute__((ext_vector_type(8)))  float          v8f;
typedef __attribute__((ext_vector_type(4)))  float          v4f;
typedef __attribute__((ext_vector_type(16))) unsigned short v16us;
typedef __attribute__((ext_vector_type(8)))  unsigned short v8us;
typedef __attribute__((ext_vector_type(4)))  unsigned short v4us;

__device__ __forceinline__ unsigned short f2bf(float f) {
  unsigned int u = __builtin_bit_cast(unsigned int, f);
  u += 0x7FFFu + ((u >> 16) & 1u);          // round-to-nearest-even
  return (unsigned short)(u >> 16);
}

__device__ __forceinline__ v16bf mk16(v8us p0, v8us p1) {
  v16us u = __builtin_shufflevector(p0, p1, 0,1,2,3,4,5,6,7,8,9,10,11,12,13,14,15);
  return __builtin_bit_cast(v16bf, u);
}

// ---------------------------------------------------------------------------
// Stage 0: one streaming pass converting x,y -> XYb (20480x512 bf16) and
// W -> Wb (224x512 bf16, rows >= 200 zero). Every element converted once.
// ---------------------------------------------------------------------------
#define XYTOT (ROWS * DIN)            // 10,485,760 f32
#define WTOT  (EP * DIN)              // 114,688 bf16 out
#define CVT_BLOCKS ((XYTOT + WTOT) / 4 / 256)   // 10352, exact

__global__ __launch_bounds__(256) void cvt_kernel(
    const float* __restrict__ x, const float* __restrict__ y,
    const float* __restrict__ W,
    unsigned short* __restrict__ XYb, unsigned short* __restrict__ Wb)
{
  const long long idx4 = ((long long)blockIdx.x * 256 + threadIdx.x) * 4;
  if (idx4 < (long long)XYTOT) {
    const long long xb = (long long)XROWS * DIN;
    const float* src = (idx4 < xb) ? (x + idx4) : (y + (idx4 - xb));
    v4f f = *(const v4f*)src;
    v4us o;
#pragma unroll
    for (int s = 0; s < 4; ++s) o[s] = f2bf(f[s]);
    *(v4us*)(XYb + idx4) = o;
  } else {
    const long long wi = idx4 - XYTOT;          // within 224 x 512
    const int e = (int)(wi >> 9);               // / 512
    v4us o = (v4us){0, 0, 0, 0};
    if (e < EDIM) {
      v4f f = *(const v4f*)(W + wi);
#pragma unroll
      for (int s = 0; s < 4; ++s) o[s] = f2bf(f[s]);
    }
    *(v4us*)(Wb + wi) = o;
  }
}

// ---------------------------------------------------------------------------
// Stage 1: pure-bf16 projection GEMM. One wave per (m-tile, n-half): A frag
// loaded once per K-step and reused across 7 n-tiles -> 7 WMMAs per step,
// 112 per wave. No conversions in the loop. Output padded to EP cols.
// ---------------------------------------------------------------------------
__global__ __launch_bounds__(256) void proj_kernel(
    const unsigned short* __restrict__ XYb, const unsigned short* __restrict__ Wb,
    const float* __restrict__ bias, unsigned short* __restrict__ P)
{
  const int wave = threadIdx.x >> 5;
  const int lane = threadIdx.x & 31;
  const int lo   = lane & 15;
  const int hi   = lane >> 4;

  const int tile = blockIdx.x * 8 + wave;       // 320 blocks * 8 = 2560
  const int mt   = tile >> 1;                   // 1280 m-tiles
  const int half = tile & 1;                    // cols [half*112, half*112+112)
  const int m0   = mt * 16;
  const int nb   = half * 112;

  const unsigned short* arow = XYb + (size_t)(m0 + lo) * DIN;
  const unsigned short* wrow = Wb + (size_t)(nb + lo) * DIN;

  v8f acc[7] = {};
  for (int k0 = 0; k0 < DIN; k0 += 32) {
    const int kA = k0 + hi * 16;                // A: K {0..15}|{16..31}
    v16bf af = mk16(*(const v8us*)(arow + kA),
                    *(const v8us*)(arow + kA + 8));
    const int kB = k0 + hi * 16;                // B: K {0..15}|{16..31}
#pragma unroll
    for (int nt = 0; nt < 7; ++nt) {
      const unsigned short* wp = wrow + (size_t)nt * 16 * DIN + kB;
      v16bf bf_ = mk16(*(const v8us*)wp, *(const v8us*)(wp + 8));
      acc[nt] = __builtin_amdgcn_wmma_f32_16x16x32_bf16(
          false, af, false, bf_, (short)0, acc[nt], false, false);
    }
  }

#pragma unroll
  for (int nt = 0; nt < 7; ++nt) {
    const int   col = nb + nt * 16 + lo;
    const float bb  = (col < EDIM) ? bias[col] : 0.0f;
#pragma unroll
    for (int r = 0; r < 8; ++r) {
      const int ro = m0 + r + hi * 8;           // C: M = r + hi*8, N = lo
      P[(size_t)ro * EP + col] = f2bf(acc[nt][r] + bb);
    }
  }
}

// ---------------------------------------------------------------------------
// Stage 2: one workgroup per (i, j-group-of-4). xp[i] (256 rows) and yp for
// 4 consecutive j (256 rows) staged via async global->LDS loads (240KB of the
// 320KB WGP LDS; rows padded to 232 elems -> conflict-free ds_load_b128).
// 8 waves compute 2x4 WMMA tiles per j; max/mean epilogue fused.
// ---------------------------------------------------------------------------
#define SROWS 256
#define SBUF_US (SROWS * LP)                    // 59392 ushorts = 118784 B
#define CPR (EP / 8)                            // 28 x 16B chunks per row
#define SMEM_BYTES (2 * SBUF_US * 2 + 8 * 64 * 4 + 64 * 4)   // 239872 B

__device__ __forceinline__ void stage_rows(const unsigned short* __restrict__ gbase,
                                           unsigned short* __restrict__ lbase,
                                           int tid) {
#if USE_ASYNC_LDS
  const unsigned long long ga = (unsigned long long)(uintptr_t)gbase;
  const unsigned lb = (unsigned)(uintptr_t)lbase;   // low 32 bits = LDS byte addr
#endif
#pragma unroll
  for (int t = 0; t < (SROWS * CPR) / 256; ++t) {   // 28 iters
    const int c   = tid + t * 256;
    const int row = c / CPR;
    const int ch  = c - row * CPR;
#if USE_ASYNC_LDS
    unsigned goff = (unsigned)(row * (EP * 2) + ch * 16);
    unsigned loff = lb + (unsigned)(row * (LP * 2) + ch * 16);
    asm volatile("global_load_async_to_lds_b128 %0, %1, %2"
                 :: "v"(loff), "v"(goff), "s"(ga) : "memory");
#else
    *(v8us*)(lbase + row * LP + ch * 8) = *(const v8us*)(gbase + row * EP + ch * 8);
#endif
  }
}

__global__ __launch_bounds__(256) void misa_kernel(
    const unsigned short* __restrict__ P, float* __restrict__ S)
{
  extern __shared__ char smem[];
  unsigned short* sX     = (unsigned short*)smem;              // xp[i]: 256 x LP
  unsigned short* sY     = sX + SBUF_US;                       // yp 4 j's: 256 x LP
  float*          swpart = (float*)(sY + SBUF_US);             // [8 waves][64 cols]
  float*          smax   = swpart + 8 * 64;                    // [64 cols]

  const int i   = blockIdx.y;
  const int jg  = blockIdx.x;                                  // group of 4 j's
  const int tid = threadIdx.x;

  stage_rows(P + (size_t)i * 256 * EP, sX, tid);
  stage_rows(P + (size_t)(XROWS + jg * 256) * EP, sY, tid);
#if USE_ASYNC_LDS
  asm volatile("s_wait_asynccnt 0x0" ::: "memory");
#endif
  __syncthreads();

  const int wave = tid >> 5;
  const int lane = tid & 31;
  const int lo   = lane & 15;
  const int hi   = lane >> 4;

#pragma unroll 1
  for (int jj = 0; jj < 4; ++jj) {
    v8f acc[2][4] = {};

#pragma unroll
    for (int kk = 0; kk < EP / 32; ++kk) {          // 7 K-steps
      const int k0 = kk * 32;
      v16bf afrag[2];
#pragma unroll
      for (int mtl = 0; mtl < 2; ++mtl) {
        const int row = (wave * 2 + mtl) * 16 + lo;
        const int off = row * LP + k0 + hi * 16;
        afrag[mtl] = mk16(*(const v8us*)(sX + off),
                          *(const v8us*)(sX + off + 8));
      }
      v16bf bfrag[4];
#pragma unroll
      for (int ntl = 0; ntl < 4; ++ntl) {
        const int t   = jj * 64 + ntl * 16 + lo;    // matchmap column = yp row
        const int off = t * LP + k0 + hi * 16;
        bfrag[ntl] = mk16(*(const v8us*)(sY + off),
                          *(const v8us*)(sY + off + 8));
      }
#pragma unroll
      for (int mtl = 0; mtl < 2; ++mtl)
#pragma unroll
        for (int ntl = 0; ntl < 4; ++ntl)
          acc[mtl][ntl] = __builtin_amdgcn_wmma_f32_16x16x32_bf16(
              false, afrag[mtl], false, bfrag[ntl], (short)0, acc[mtl][ntl],
              false, false);
    }

    // ---- fused epilogue: max over 256 rows per column, then mean over 64
#pragma unroll
    for (int ntl = 0; ntl < 4; ++ntl) {
      float m = acc[0][ntl][0];
#pragma unroll
      for (int mtl = 0; mtl < 2; ++mtl)
#pragma unroll
        for (int r = 0; r < 8; ++r)
          m = fmaxf(m, acc[mtl][ntl][r]);
      m = fmaxf(m, __shfl_xor(m, 16, 32));          // merge lane row-halves
      if (hi == 0) swpart[wave * 64 + ntl * 16 + lo] = m;
    }
    __syncthreads();

    if (tid < 64) {
      float m = swpart[tid];
#pragma unroll
      for (int w = 1; w < 8; ++w) m = fmaxf(m, swpart[w * 64 + tid]);
      smax[tid] = m;
    }
    __syncthreads();

    if (tid == 0) {
      float s = 0.0f;
      for (int t = 0; t < 64; ++t) s += smax[t];
      S[i * 64 + jg * 4 + jj] = s * (1.0f / 64.0f);
    }
    __syncthreads();                                // protect swpart/smax reuse
  }
}

// ---------------------------------------------------------------------------
extern "C" void kernel_launch(void* const* d_in, const int* in_sizes, int n_in,
                              void* d_out, int out_size, void* d_ws, size_t ws_size,
                              hipStream_t stream) {
  (void)in_sizes; (void)n_in; (void)out_size; (void)ws_size;
  const float* x = (const float*)d_in[0];
  const float* y = (const float*)d_in[1];
  const float* W = (const float*)d_in[2];
  const float* b = (const float*)d_in[3];

  unsigned short* XYb = (unsigned short*)d_ws;        // 20480 x 512 bf16
  unsigned short* Wb  = XYb + (size_t)ROWS * DIN;     // 224 x 512 bf16
  unsigned short* P   = Wb + (size_t)EP * DIN;        // 20480 x 224 bf16
  float* S = (float*)d_out;                           // 64 x 64 f32

  // 240KB dynamic LDS opt-in (CDNA5 WGP has 320KB)
  hipFuncSetAttribute(reinterpret_cast<const void*>(misa_kernel),
                      hipFuncAttributeMaxDynamicSharedMemorySize, SMEM_BYTES);

  cvt_kernel<<<CVT_BLOCKS, 256, 0, stream>>>(x, y, W, XYb, Wb);
  proj_kernel<<<320, 256, 0, stream>>>(XYb, Wb, b, P);
  misa_kernel<<<dim3(16, 64), 256, SMEM_BYTES, stream>>>(P, S);
}
